// CostVolumeModulation_1331439862521
// MI455X (gfx1250) — matrix operand order repaired and checked by reading, not compiled, not run on hardware
//
#include <hip/hip_runtime.h>
#include <hip/hip_bf16.h>

typedef _Float16 half_t;
typedef __attribute__((ext_vector_type(16))) _Float16 v16h;
typedef __attribute__((ext_vector_type(8)))  float    v8f;

union FragU { uint4 u[2]; v16h h; };
union PackU { uint4 u; half_t h[8]; };

static __device__ __forceinline__ float lrelu_f(float x) {
    return x >= 0.f ? x : 0.1f * x;
}

// ---------------------------------------------------------------------------
// Pack conv weights (f32 OIHW) into CDNA5 WMMA A-fragment layout, f16.
// Layout: [octile][tap][icchunk][lane 0..31][16 halfs]
// A order used here (contiguous k, B matches): lanes 0-15 hold M=lane, elems 0..15 -> K 0..15
//                     lanes 16-31 hold M=lane-16, elems 0..15 -> K 16..31
// ---------------------------------------------------------------------------
__global__ void pack_weights(const float* __restrict__ w, half_t* __restrict__ out,
                             int OC, int Cin, int cinPad, int ntaps, int octiles)
{
    int t = blockIdx.x * blockDim.x + threadIdx.x;
    int nch = cinPad >> 5;
    int total = octiles * ntaps * nch * 32;
    if (t >= total) return;
    int lane = t & 31;
    int rest = t >> 5;
    int chunk = rest % nch; rest /= nch;
    int tap   = rest % ntaps; rest /= ntaps;
    int octile = rest;
    int oc = octile * 16 + (lane & 15);
    int hi = (lane >= 16) ? 16 : 0;
    half_t* dst = out + (size_t)t * 16;
    #pragma unroll
    for (int e = 0; e < 16; ++e) {
        int K = e + hi;
        int ic = chunk * 32 + K;
        float v = 0.f;
        if (oc < OC && ic < Cin) v = w[((size_t)oc * Cin + ic) * ntaps + tap];
        dst[e] = (half_t)v;
    }
}

// ---------------------------------------------------------------------------
// Bilinear warp of f2 by flow*MULT with validity mask; output channel-last f32.
// ---------------------------------------------------------------------------
__global__ void warp_kernel(const float* __restrict__ feats, const float* __restrict__ flow,
                            float* __restrict__ f2w, int B, int C, int H, int W)
{
    int t = blockIdx.x * blockDim.x + threadIdx.x;
    int total = B * H * W;
    if (t >= total) return;
    int x = t % W; int y = (t / W) % H; int b = t / (W * H);
    const float mult = 2.5f;  // DIV_FLOW / 2^(NUM_LEVELS-LEVEL+1) = 20/8
    float fx = flow[((size_t)(b * 2 + 0) * H + y) * W + x];
    float fy = flow[((size_t)(b * 2 + 1) * H + y) * W + x];
    float px = (float)x + fx * mult;
    float py = (float)y + fy * mult;
    float x0f = floorf(px), y0f = floorf(py);
    float wx1 = px - x0f, wy1 = py - y0f;
    int xi = (int)x0f, yi = (int)y0f;
    float wgt[4] = { (1.f - wy1) * (1.f - wx1), (1.f - wy1) * wx1,
                     wy1 * (1.f - wx1),         wy1 * wx1 };
    int xs[4] = { xi, xi + 1, xi,     xi + 1 };
    int ys[4] = { yi, yi,     yi + 1, yi + 1 };
    int off[4]; float we[4]; float wsum = 0.f;
    #pragma unroll
    for (int k = 0; k < 4; ++k) {
        bool valid = (xs[k] >= 0) && (xs[k] < W) && (ys[k] >= 0) && (ys[k] < H);
        int xc = xs[k] < 0 ? 0 : (xs[k] > W - 1 ? W - 1 : xs[k]);
        int yc = ys[k] < 0 ? 0 : (ys[k] > H - 1 ? H - 1 : ys[k]);
        off[k] = yc * W + xc;
        we[k] = valid ? wgt[k] : 0.f;
        wsum += we[k];
    }
    float mask = (wsum >= 1.0f) ? 1.f : 0.f;
    const float* f2 = feats + (size_t)(b * 2 + 1) * C * H * W;
    float* dst = f2w + (size_t)t * C;
    for (int c = 0; c < C; ++c) {
        const float* fc = f2 + (size_t)c * H * W;
        float v = we[0] * fc[off[0]] + we[1] * fc[off[1]] +
                  we[2] * fc[off[2]] + we[3] * fc[off[3]];
        dst[c] = v * mask;
    }
}

// ---------------------------------------------------------------------------
// Build channel-last conv input x0 (f16, 192-ch padded): ch 0..95 = f1,
// ch 177 = conf, ch 178..191 = 0 (corr fills 96..176). Also f1 channel-last f32.
// ---------------------------------------------------------------------------
__global__ void build_x0(const float* __restrict__ feats, const float* __restrict__ conf,
                         half_t* __restrict__ x0, float* __restrict__ f1cl,
                         int B, int C, int H, int W, int CPAD)
{
    int t = blockIdx.x * blockDim.x + threadIdx.x;
    int total = B * H * W;
    if (t >= total) return;
    int x = t % W; int y = (t / W) % H; int b = t / (W * H);
    const float* f1 = feats + (size_t)(b * 2) * C * H * W;
    half_t* dx = x0 + (size_t)t * CPAD;
    float* df = f1cl + (size_t)t * C;
    int sp = y * W + x;
    for (int c = 0; c < C; ++c) {
        float v = f1[(size_t)c * H * W + sp];
        df[c] = v;
        dx[c] = (half_t)v;
    }
    dx[177] = (half_t)conf[(size_t)(b * H + y) * W + x];
    for (int c = 178; c < CPAD; ++c) dx[c] = (half_t)0.f;
}

// ---------------------------------------------------------------------------
// 9x9 patch correlation: corr[d] = lrelu(sum_c f1*f2w_shift)/96.
// Writes f32 NCHW corr (for final fuse) and f16 into x0 channels 96..176.
// ---------------------------------------------------------------------------
__global__ void corr_kernel(const float* __restrict__ f1cl, const float* __restrict__ f2w,
                            float* __restrict__ corr, half_t* __restrict__ x0,
                            int B, int C, int H, int W, int CPAD)
{
    int t = blockIdx.x * blockDim.x + threadIdx.x;
    int total = B * H * W;
    if (t >= total) return;
    int x = t % W; int y = (t / W) % H; int b = t / (W * H);
    const float4* a4 = reinterpret_cast<const float4*>(f1cl + (size_t)t * C);
    float4 areg[24];
    #pragma unroll
    for (int i = 0; i < 24; ++i) areg[i] = a4[i];
    for (int d = 0; d < 81; ++d) {
        int yy = y + d / 9 - 4;
        int xx = x + d % 9 - 4;
        float s = 0.f;
        if (yy >= 0 && yy < H && xx >= 0 && xx < W) {
            const float4* b4 = reinterpret_cast<const float4*>(
                f2w + ((size_t)(b * H + yy) * W + xx) * C);
            #pragma unroll
            for (int i = 0; i < 24; ++i) {
                float4 q = b4[i];
                s += areg[i].x * q.x + areg[i].y * q.y + areg[i].z * q.z + areg[i].w * q.w;
            }
        }
        float r = lrelu_f(s) * (1.f / 96.f);
        corr[((size_t)(b * 81 + d) * H + y) * W + x] = r;
        x0[(size_t)t * CPAD + 96 + d] = (half_t)r;
    }
}

// ---------------------------------------------------------------------------
// Implicit-GEMM 3x3 conv (pad 1) via v_wmma_f32_16x16x32_f16.
// Input channel-last f16 (CIN_PAD mult of 32); output channel-last f16 + lrelu.
// 8 waves/WG; each wave computes OCW oc x 16 px (OCW/16 accumulators) so one
// LDS B-fragment feeds NACC WMMAs.
// Double-buffered halo tiles (3 x (NT+2) x 32ch) filled with CDNA5 async
// copies (GLOBAL_LOAD_ASYNC_TO_LDS_B128): chunk k+1's DMA overlaps chunk k's
// WMMAs; fence is s_wait_asynccnt 0 + barrier. Zero-pad border slots are
// written once (async loads never touch them, so zeros persist).
// ---------------------------------------------------------------------------
template<int CIN_PAD, int WG_OC, int OCW, int NT>
__global__ void __launch_bounds__(256) conv3x3_wmma(
    const half_t* __restrict__ xin, const half_t* __restrict__ wpk,
    const float* __restrict__ bias, half_t* __restrict__ xout,
    int OCtot, int H, int W)
{
    constexpr int NCH = CIN_PAD / 32;
    constexpr int NACC = OCW / 16;
    constexpr int NOC_SUB = WG_OC / OCW;
    constexpr int NPX_SUB = 8 / NOC_SUB;
    static_assert(NPX_SUB * 16 == NT, "tile shape mismatch");
    constexpr int LW = NT + 2;
    constexpr int SLOTS = 3 * LW * 4;      // uint4 slots per buffer
    __shared__ uint4 lds4[2][SLOTS];

    int tid = threadIdx.x;
    int lane = tid & 31;
    int wave = tid >> 5;
    int ocsub = wave % NOC_SUB;
    int pxsub = wave / NOC_SUB;

    int xtiles = W / NT;
    int y = blockIdx.x / xtiles;
    int xbase = (blockIdx.x % xtiles) * NT;
    int b = blockIdx.z;
    int ot0 = blockIdx.y * (WG_OC / 16) + ocsub * NACC;

    int hi = (lane >= 16);
    int px = pxsub * 16 + (lane & 15);
    int xg = xbase + px;

    // warm GL2 with this block's weight stream (global_prefetch_b8)
    __builtin_prefetch(wpk + (size_t)ot0 * 9 * NCH * 512, 0, 1);

    // zero-pad border slots once, in both buffers (async never writes them)
    for (int i = tid; i < SLOTS; i += 256) {
        int pos = i >> 2;
        int r  = pos / LW;
        int xx = pos % LW;
        int gy = y + r - 1;
        int gx = xbase + xx - 1;
        if (!(gy >= 0 && gy < H && gx >= 0 && gx < W)) {
            lds4[0][i] = make_uint4(0u, 0u, 0u, 0u);
            lds4[1][i] = make_uint4(0u, 0u, 0u, 0u);
        }
    }

    // issue async DMA of one 3 x LW x 32ch chunk into buffer `buf`
    auto issue_chunk = [&](int chunk, int buf) {
        for (int i = tid; i < SLOTS; i += 256) {
            int seg = i & 3;
            int pos = i >> 2;
            int r  = pos / LW;
            int xx = pos % LW;
            int gy = y + r - 1;
            int gx = xbase + xx - 1;
            if (gy >= 0 && gy < H && gx >= 0 && gx < W) {
                unsigned lds_off = (unsigned)(unsigned long long)(&lds4[buf][i]);
                unsigned long long gp = (unsigned long long)(
                    xin + ((size_t)(b * H + gy) * W + gx) * CIN_PAD + chunk * 32 + seg * 8);
                asm volatile("global_load_async_to_lds_b128 %0, %1, off"
                             :: "v"(lds_off), "v"(gp) : "memory");
            }
        }
    };

    v8f acc[NACC] = {};

    issue_chunk(0, 0);
    asm volatile("s_wait_asynccnt 0" ::: "memory");
    __syncthreads();

    for (int chunk = 0; chunk < NCH; ++chunk) {
        int cur = chunk & 1;
        if (chunk + 1 < NCH) issue_chunk(chunk + 1, (chunk + 1) & 1);

        #pragma unroll
        for (int ky = 0; ky < 3; ++ky) {
            #pragma unroll
            for (int kx = 0; kx < 3; ++kx) {
                int tap = ky * 3 + kx;
                // B fragment (shared across the NACC oc-tiles of this wave)
                FragU bf;
                int pos = ky * LW + (px + kx);
                const uint4* bl = &lds4[cur][pos * 4 + (hi ? 2 : 0)];
                bf.u[0] = bl[0]; bf.u[1] = bl[1];
                #pragma unroll
                for (int a = 0; a < NACC; ++a) {
                    FragU af;
                    const uint4* aw = reinterpret_cast<const uint4*>(
                        wpk + (((size_t)(ot0 + a) * 9 + tap) * NCH + chunk) * 512 + lane * 16);
                    af.u[0] = aw[0]; af.u[1] = aw[1];
                    acc[a] = __builtin_amdgcn_wmma_f32_16x16x32_f16(
                        false, af.h, false, bf.h, (short)0, acc[a], false, false);
                }
            }
        }

        asm volatile("s_wait_asynccnt 0" ::: "memory");
        __syncthreads();
    }

    // epilogue: bias + lrelu, pack 8 contiguous channels -> one b128 store each
    #pragma unroll
    for (int a = 0; a < NACC; ++a) {
        int ocb = (ot0 + a) * 16 + (hi ? 8 : 0);
        float4 bA = reinterpret_cast<const float4*>(bias + ocb)[0];
        float4 bB = reinterpret_cast<const float4*>(bias + ocb)[1];
        float bb[8] = {bA.x, bA.y, bA.z, bA.w, bB.x, bB.y, bB.z, bB.w};
        PackU p;
        #pragma unroll
        for (int r = 0; r < 8; ++r) {
            float v = acc[a][r] + bb[r];
            p.h[r] = (half_t)(v >= 0.f ? v : 0.1f * v);
        }
        uint4* dst = reinterpret_cast<uint4*>(
            xout + ((size_t)(b * H + y) * W + xg) * OCtot + ocb);
        dst[0] = p.u;
    }
}

// ---------------------------------------------------------------------------
// Fused 1x1 convs (32->81, padded to 96 rows) for scalar & offset branches via
// WMMA (single K=32 per branch), then out = (s2+bs2)*corr + (o2+bo2).
// 6 waves per block, wave = oc-tile, 16 px per block. The oc<81 tail guard is
// wave-uniform (only oc-tile 5 is partial) to avoid per-lane EXEC churn.
// ---------------------------------------------------------------------------
__global__ void __launch_bounds__(192) final_mod(
    const half_t* __restrict__ s1, const half_t* __restrict__ o1,
    const half_t* __restrict__ wps, const half_t* __restrict__ wpo,
    const float* __restrict__ bs2, const float* __restrict__ bo2,
    const float* __restrict__ corr, float* __restrict__ out,
    int H, int W)
{
    int tid = threadIdx.x;
    int lane = tid & 31;
    int octile = tid >> 5;          // 0..5 -> oc 0..95 (only 0..80 stored)
    int xtiles = W >> 4;
    int y = blockIdx.x / xtiles;
    int x = (blockIdx.x % xtiles) * 16 + (lane & 15);
    int b = blockIdx.z;
    int hi = (lane >= 16);

    size_t pixBase = ((size_t)(b * H + y) * W + x) * 32 + (hi ? 16 : 0);
    FragU bS, bO;
    const uint4* ps = reinterpret_cast<const uint4*>(s1 + pixBase);
    bS.u[0] = ps[0]; bS.u[1] = ps[1];
    const uint4* po = reinterpret_cast<const uint4*>(o1 + pixBase);
    bO.u[0] = po[0]; bO.u[1] = po[1];

    FragU aS, aO;
    const uint4* pas = reinterpret_cast<const uint4*>(wps + (size_t)octile * 512 + lane * 16);
    aS.u[0] = pas[0]; aS.u[1] = pas[1];
    const uint4* pao = reinterpret_cast<const uint4*>(wpo + (size_t)octile * 512 + lane * 16);
    aO.u[0] = pao[0]; aO.u[1] = pao[1];

    v8f accS = {0.f, 0.f, 0.f, 0.f, 0.f, 0.f, 0.f, 0.f};
    v8f accO = {0.f, 0.f, 0.f, 0.f, 0.f, 0.f, 0.f, 0.f};
    accS = __builtin_amdgcn_wmma_f32_16x16x32_f16(false, aS.h, false, bS.h, (short)0, accS, false, false);
    accO = __builtin_amdgcn_wmma_f32_16x16x32_f16(false, aO.h, false, bO.h, (short)0, accO, false, false);

    int ocb = octile * 16 + (hi ? 8 : 0);
    if (octile < 5) {               // wave-uniform: all 8 channels valid
        #pragma unroll
        for (int r = 0; r < 8; ++r) {
            int oc = ocb + r;
            size_t idx = ((size_t)(b * 81 + oc) * H + y) * W + x;
            out[idx] = (accS[r] + bs2[oc]) * corr[idx] + (accO[r] + bo2[oc]);
        }
    } else {                        // oc-tile 5: only oc==80 exists (r==0, lo half)
        if (!hi) {
            size_t idx = ((size_t)(b * 81 + 80) * H + y) * W + x;
            out[idx] = (accS[0] + bs2[80]) * corr[idx] + (accO[0] + bo2[80]);
        }
    }
}

// ---------------------------------------------------------------------------
extern "C" void kernel_launch(void* const* d_in, const int* in_sizes, int n_in,
                              void* d_out, int out_size, void* d_ws, size_t ws_size,
                              hipStream_t stream)
{
    const int B = 4, C = 96, H = 128, W = 256;
    const float* feats = (const float*)d_in[0];
    const float* flow  = (const float*)d_in[1];
    const float* conf  = (const float*)d_in[2];
    const float* w1  = (const float*)d_in[3];  const float* b1  = (const float*)d_in[4];
    const float* w2  = (const float*)d_in[5];  const float* b2  = (const float*)d_in[6];
    const float* ws1 = (const float*)d_in[7];  const float* bs1 = (const float*)d_in[8];
    const float* ws2 = (const float*)d_in[9];  const float* bs2 = (const float*)d_in[10];
    const float* wo1 = (const float*)d_in[11]; const float* bo1 = (const float*)d_in[12];
    const float* wo2 = (const float*)d_in[13]; const float* bo2 = (const float*)d_in[14];
    float* out = (float*)d_out;

    char* ws = (char*)d_ws;
    size_t off = 0;
    auto take = [&](size_t bytes) -> char* {
        char* p = ws + off;
        off = (off + bytes + 255) & ~(size_t)255;
        return p;
    };
    const size_t NPIX = (size_t)B * H * W;
    float*  f2w  = (float*)  take(NPIX * 96 * 4);
    float*  f1cl = (float*)  take(NPIX * 96 * 4);
    half_t* x0   = (half_t*) take(NPIX * 192 * 2);
    float*  corr = (float*)  take(NPIX * 81 * 4);
    half_t* x1   = (half_t*) take(NPIX * 128 * 2);
    half_t* x2   = (half_t*) take(NPIX * 64 * 2);
    half_t* s1b  = (half_t*) take(NPIX * 32 * 2);
    half_t* o1b  = (half_t*) take(NPIX * 32 * 2);
    half_t* w1p  = (half_t*) take((size_t)8 * 9 * 6 * 512 * 2);
    half_t* w2p  = (half_t*) take((size_t)4 * 9 * 4 * 512 * 2);
    half_t* ws1p = (half_t*) take((size_t)2 * 9 * 2 * 512 * 2);
    half_t* wo1p = (half_t*) take((size_t)2 * 9 * 2 * 512 * 2);
    half_t* ws2p = (half_t*) take((size_t)6 * 512 * 2);
    half_t* wo2p = (half_t*) take((size_t)6 * 512 * 2);

    auto cdiv = [](int a, int b) { return (a + b - 1) / b; };

    // weight prepack into WMMA A-fragment layout
    pack_weights<<<cdiv(8 * 9 * 6 * 32, 256), 256, 0, stream>>>(w1,  w1p,  128, 178, 192, 9, 8);
    pack_weights<<<cdiv(4 * 9 * 4 * 32, 256), 256, 0, stream>>>(w2,  w2p,   64, 128, 128, 9, 4);
    pack_weights<<<cdiv(2 * 9 * 2 * 32, 256), 256, 0, stream>>>(ws1, ws1p,  32,  64,  64, 9, 2);
    pack_weights<<<cdiv(2 * 9 * 2 * 32, 256), 256, 0, stream>>>(wo1, wo1p,  32,  64,  64, 9, 2);
    pack_weights<<<cdiv(6 * 32, 256),          256, 0, stream>>>(ws2, ws2p,  81,  32,  32, 1, 6);
    pack_weights<<<cdiv(6 * 32, 256),          256, 0, stream>>>(wo2, wo2p,  81,  32,  32, 1, 6);

    // warp + correlation + input assembly
    warp_kernel<<<cdiv(B * H * W, 256), 256, 0, stream>>>(feats, flow, f2w, B, C, H, W);
    build_x0 <<<cdiv(B * H * W, 256), 256, 0, stream>>>(feats, conf, x0, f1cl, B, C, H, W, 192);
    corr_kernel<<<cdiv(B * H * W, 256), 256, 0, stream>>>(f1cl, f2w, corr, x0, B, C, H, W, 192);

    // WMMA conv stack (each wave: 2 oc-tiles x 16 px; async double-buffered LDS)
    conv3x3_wmma<192, 64, 32,  64><<<dim3((W /  64) * H, 2, B), 256, 0, stream>>>(x0, w1p,  b1,  x1,  128, H, W);
    conv3x3_wmma<128, 64, 32,  64><<<dim3((W /  64) * H, 1, B), 256, 0, stream>>>(x1, w2p,  b2,  x2,   64, H, W);
    conv3x3_wmma< 64, 32, 32, 128><<<dim3((W / 128) * H, 1, B), 256, 0, stream>>>(x2, ws1p, bs1, s1b,  32, H, W);
    conv3x3_wmma< 64, 32, 32, 128><<<dim3((W / 128) * H, 1, B), 256, 0, stream>>>(x2, wo1p, bo1, o1b,  32, H, W);

    // fused 1x1 convs + modulation
    final_mod<<<dim3((W / 16) * H, 1, B), 192, 0, stream>>>(s1b, o1b, ws2p, wo2p, bs2, bo2, corr, out, H, W);
}
